// CLIPViTFM2_41016937677170
// MI455X (gfx1250) — matrix operand order repaired and checked by reading, not compiled, not run on hardware
//
#include <hip/hip_runtime.h>

typedef __attribute__((ext_vector_type(16))) _Float16 v16h;
typedef __attribute__((ext_vector_type(8)))  _Float16 v8h;
typedef __attribute__((ext_vector_type(8)))  float    v8f;

namespace {
constexpr int Dm   = 768;
constexpr int Hh   = 12;
constexpr int HDh  = 64;
constexpr int Ll   = 196;   // GRID*GRID
constexpr int Ss   = 197;   // L + 1
constexpr int SPp  = 224;   // padded key dim for scores/probs (multiple of 32)
constexpr int KTP  = 256;   // padded key dim for K^T tiles (multiple of 64)
constexpr int FFm  = 3072;
constexpr int OUTm = 512;
constexpr int Nb   = 16;    // batch per image set
constexpr int BFULL= 32;    // concat batch for first 10 layers
}

// ---------------------------------------------------------------------------
// WMMA GEMM (NN form only):
//   C[M,N] = act( alpha * A[M,K] x B[K,N] + bias + resid )
// 256 threads = 8 waves; block tile 128(M) x 64(N); each wave owns 32x32
// (2 A-frags x 2 B-frags -> 4 v_wmma_f32_16x16x32_f16 per 32-wide k-step).
// Requirements: K % 32 == 0; B rows are readable up to col roundup(N,64)
// (all our B buffers are padded accordingly). A rows are clamped to M-1
// (branch-free loads; out-of-range rows discarded at store).
// Batched over blockIdx.z: (batch, head) = (z / headsPerZ, z % headsPerZ).
// ---------------------------------------------------------------------------
__global__ __launch_bounds__(256)
void gemm_wmma_kernel(const _Float16* __restrict__ A, long long lda,
                      const _Float16* __restrict__ B, long long ldb,
                      int M, int N, int K,
                      const float* __restrict__ bias,
                      const float* __restrict__ resid, long long ldr,
                      float alpha, int act,
                      float* __restrict__ C32, long long ldc32,
                      _Float16* __restrict__ C16, long long ldc16,
                      int headsPerZ,
                      long long sAb, long long sAh,
                      long long sBb, long long sBh,
                      long long sC32b, long long sC32h,
                      long long sC16b, long long sC16h)
{
    const int z  = blockIdx.z;
    const int bz = z / headsPerZ;
    const int hz = z - bz * headsPerZ;
    A += (long long)bz * sAb + (long long)hz * sAh;
    B += (long long)bz * sBb + (long long)hz * sBh;
    const long long c32off = (long long)bz * sC32b + (long long)hz * sC32h;
    const long long c16off = (long long)bz * sC16b + (long long)hz * sC16h;

    const int lane = threadIdx.x & 31;
    const int wv   = threadIdx.x >> 5;                 // 8 waves
    const int tm   = blockIdx.y * 128 + (wv >> 1) * 32; // 4 wave-rows of 32
    const int tn   = blockIdx.x * 64  + (wv & 1) * 32;  // 2 wave-cols of 32
    const int hi   = lane >> 4;
    const int lm   = lane & 15;

    // Branch-free A addressing: clamp rows into [0, M-1].
    int r0 = tm + lm;        if (r0 > M - 1) r0 = M - 1;
    int r1 = tm + 16 + lm;   if (r1 > M - 1) r1 = M - 1;
    const _Float16* pa0 = A + (long long)r0 * lda + hi * 8;
    const _Float16* pa1 = A + (long long)r1 * lda + hi * 8;
    const _Float16* pb  = B + tn + lm;

    v8f c00 = {0.f,0.f,0.f,0.f,0.f,0.f,0.f,0.f};
    v8f c01 = c00, c10 = c00, c11 = c00;

    for (int k0 = 0; k0 < K; k0 += 32) {
        // A fragments (16-bit A 16x32 layout: lane half selects K sub-block)
        v16h a0, a1;
        {
            v8h lo = *(const v8h*)(pa0 + k0);
            v8h hv = *(const v8h*)(pa0 + k0 + 16);
#pragma unroll
            for (int e = 0; e < 8; ++e) { a0[e] = lo[e]; a0[8 + e] = hv[e]; }
        }
        {
            v8h lo = *(const v8h*)(pa1 + k0);
            v8h hv = *(const v8h*)(pa1 + k0 + 16);
#pragma unroll
            for (int e = 0; e < 8; ++e) { a1[e] = lo[e]; a1[8 + e] = hv[e]; }
        }
        // B fragments: lane = column (tn+lm), element = K row (same order as A)
        v16h b0, b1;
        {
            const _Float16* pbk = pb + (long long)k0 * ldb;
            // prefetch next k-tile of B (global_prefetch_b8, speculative)
            __builtin_prefetch((const void*)(pbk + 32 * ldb), 0, 1);
#pragma unroll
            for (int e = 0; e < 8; ++e) {
                const long long rl = (long long)(hi * 8 + e) * ldb;
                const long long rh = (long long)(16 + hi * 8 + e) * ldb;
                b0[e] = pbk[rl]; b0[8 + e] = pbk[rh];
                b1[e] = pbk[rl + 16]; b1[8 + e] = pbk[rh + 16];
            }
        }

        c00 = __builtin_amdgcn_wmma_f32_16x16x32_f16(false, a0, false, b0,
                                                     (short)0, c00, false, false);
        c01 = __builtin_amdgcn_wmma_f32_16x16x32_f16(false, a0, false, b1,
                                                     (short)0, c01, false, false);
        c10 = __builtin_amdgcn_wmma_f32_16x16x32_f16(false, a1, false, b0,
                                                     (short)0, c10, false, false);
        c11 = __builtin_amdgcn_wmma_f32_16x16x32_f16(false, a1, false, b1,
                                                     (short)0, c11, false, false);
    }

    // Epilogue. C layout: VGPR r -> row base + 8*hi + r, col = tn + lm (+16).
    const int colBase = tn + lm;
#pragma unroll
    for (int part = 0; part < 4; ++part) {
        const v8f cc = (part == 0) ? c00 : (part == 1) ? c01
                     : (part == 2) ? c10 : c11;
        const int rbase = tm + ((part >> 1) ? 16 : 0) + hi * 8;
        const int col   = colBase + ((part & 1) ? 16 : 0);
        if (col >= N) continue;
        float bv = bias ? bias[col] : 0.f;
#pragma unroll
        for (int r = 0; r < 8; ++r) {
            const int row = rbase + r;
            if (row >= M) continue;
            float v = cc[r] * alpha + bv;
            if (resid) v += resid[(long long)row * ldr + col];
            if (act == 1) v = v / (1.f + __expf(-1.702f * v)); // x*sigmoid(1.702x)
            if (C32) C32[c32off + (long long)row * ldc32 + col] = v;
            if (C16) C16[c16off + (long long)row * ldc16 + col] = (_Float16)v;
        }
    }
}

// ---------------------------------------------------------------------------
// LayerNorm: one block per row, LDS reduction. Optional f32 and f16 outputs.
// ---------------------------------------------------------------------------
__global__ __launch_bounds__(256)
void ln_kernel(const float* __restrict__ X, long long rowStride, int M, int d,
               const float* __restrict__ g, const float* __restrict__ b,
               float* __restrict__ o32, long long ld32,
               _Float16* __restrict__ o16, long long ld16)
{
    __shared__ float sred[256];
    const int row = blockIdx.x;
    if (row >= M) return;
    const float* xr = X + (long long)row * rowStride;
    const int tid = threadIdx.x;

    float s = 0.f;
    for (int j = tid; j < d; j += 256) s += xr[j];
    sred[tid] = s; __syncthreads();
    for (int st = 128; st > 0; st >>= 1) {
        if (tid < st) sred[tid] += sred[tid + st];
        __syncthreads();
    }
    const float mu = sred[0] / (float)d;
    __syncthreads();

    float s2 = 0.f;
    for (int j = tid; j < d; j += 256) { float t = xr[j] - mu; s2 += t * t; }
    sred[tid] = s2; __syncthreads();
    for (int st = 128; st > 0; st >>= 1) {
        if (tid < st) sred[tid] += sred[tid + st];
        __syncthreads();
    }
    const float rs = rsqrtf(sred[0] / (float)d + 1e-5f);

    for (int j = tid; j < d; j += 256) {
        float y = (xr[j] - mu) * rs * g[j] + b[j];
        if (o32) o32[(long long)row * ld32 + j] = y;
        if (o16) o16[(long long)row * ld16 + j] = (_Float16)y;
    }
}

// ---------------------------------------------------------------------------
// Softmax over keys (197 valid, 224 padded) -> f16 probs. Optional CLS-row
// mask bias: for q==0, key j in [1,196]: +(-1e30) where pred_mask==0.
// grid = (197, B*H)
// ---------------------------------------------------------------------------
__global__ __launch_bounds__(256)
void softmax_kernel(const float* __restrict__ scores, _Float16* __restrict__ probs,
                    const float* __restrict__ pm, int H)
{
    __shared__ float sred[256];
    const int q = blockIdx.x;
    const long long z = blockIdx.y;
    const float* srow = scores + (z * (long long)Ss + q) * SPp;
    _Float16*    prow = probs  + (z * (long long)Ss + q) * SPp;
    const int tid = threadIdx.x;

    const bool valid = tid < Ss;
    float sv = 0.f;
    if (valid) {
        sv = srow[tid];
        if (pm && q == 0 && tid >= 1) {
            const int b = (int)(z / H);
            if (pm[b * Ll + (tid - 1)] == 0.f) sv += -1e30f;
        }
    }
    sred[tid] = valid ? sv : -3.0e38f;
    __syncthreads();
    for (int st = 128; st > 0; st >>= 1) {
        if (tid < st) sred[tid] = fmaxf(sred[tid], sred[tid + st]);
        __syncthreads();
    }
    const float mx = sred[0];
    __syncthreads();

    const float e = valid ? __expf(sv - mx) : 0.f;
    sred[tid] = e; __syncthreads();
    for (int st = 128; st > 0; st >>= 1) {
        if (tid < st) sred[tid] += sred[tid + st];
        __syncthreads();
    }
    const float inv = 1.f / sred[0];
    if (tid < SPp) prow[tid] = valid ? (_Float16)(e * inv) : (_Float16)0.f;
}

// ---------------------------------------------------------------------------
// fp32 -> f16 transposed weight convert: Wt[k*N + n] = W[n*K + k]
// ---------------------------------------------------------------------------
__global__ void convert_tr_kernel(const float* __restrict__ W,
                                  _Float16* __restrict__ Wt, int Nn, int Kk)
{
    long long i = blockIdx.x * 256LL + threadIdx.x;
    long long tot = (long long)Nn * Kk;
    if (i >= tot) return;
    int k = (int)(i / Nn);
    int n = (int)(i % Nn);
    Wt[i] = (_Float16)W[(long long)n * Kk + k];
}

__global__ void convert_kernel(const float* __restrict__ W,
                               _Float16* __restrict__ o, long long tot)
{
    long long i = blockIdx.x * 256LL + threadIdx.x;
    if (i < tot) o[i] = (_Float16)W[i];
}

// ---------------------------------------------------------------------------
// K^T per head: kT[((b*H + h)*64 + d)*256 + s] = K[b, s, h*64 + d], zero pad.
// Makes the scores GEMM pure NN-form with contiguous lane loads.
// ---------------------------------------------------------------------------
__global__ void kt_fill_kernel(const _Float16* __restrict__ qkv,
                               _Float16* __restrict__ kT, int Bb)
{
    long long i = blockIdx.x * 256LL + threadIdx.x;
    const long long tot = (long long)Bb * Hh * HDh * KTP;
    if (i >= tot) return;
    const int s = (int)(i & (KTP - 1));
    const long long r = i >> 8;              // KTP == 256
    const int d = (int)(r % HDh);
    const long long zz = r / HDh;
    const int h = (int)(zz % Hh);
    const long long b = zz / Hh;
    _Float16 v = (_Float16)0.f;
    if (s < Ss)
        v = qkv[(b * Ss + s) * (3LL * Dm) + Dm + (long long)h * HDh + d];
    kT[i] = v;
}

// ---------------------------------------------------------------------------
// im2col of both image sets -> f16 patch matrix (32*196 rows x 768 cols).
// ---------------------------------------------------------------------------
__global__ void im2col_kernel(const float* __restrict__ imgA,
                              const float* __restrict__ imgB,
                              _Float16* __restrict__ out)
{
    long long i = blockIdx.x * 256LL + threadIdx.x;
    const long long tot = (long long)BFULL * Ll * Dm;
    if (i >= tot) return;
    const int col = (int)(i % Dm);
    const long long row = i / Dm;
    const int l   = (int)(row % Ll);
    const int img = (int)(row / Ll);
    const int py = l / 14, px = l % 14;
    const int c  = col >> 8, rem = col & 255;
    const int iy = rem >> 4, ix = rem & 15;
    const float* src = (img < Nb) ? imgA : imgB;
    const int n = (img < Nb) ? img : img - Nb;
    const float v = src[(((long long)n * 3 + c) * 224 + (py * 16 + iy)) * 224
                        + (px * 16 + ix)];
    out[i] = (_Float16)v;
}

// x_raw[b,s,:] = (s==0 ? class_emb : feat[b*196+s-1]) + pos_emb[s]
__global__ void assemble_kernel(const float* __restrict__ feat,
                                const float* __restrict__ cls,
                                const float* __restrict__ pos,
                                float* __restrict__ out)
{
    long long i = blockIdx.x * 256LL + threadIdx.x;
    const long long tot = (long long)BFULL * Ss * Dm;
    if (i >= tot) return;
    const int d = (int)(i % Dm);
    const long long r = i / Dm;
    const int s = (int)(r % Ss);
    const long long b = r / Ss;
    const float v = (s == 0) ? cls[d] : feat[(b * Ll + (s - 1)) * Dm + d];
    out[i] = v + pos[(long long)s * Dm + d];
}

// tmp = concat(x2_cls, x2_tok * pm) * 2 + x
__global__ void masked_combine_kernel(float* __restrict__ out,
                                      const float* __restrict__ x,
                                      const float* __restrict__ x2,
                                      const float* __restrict__ pm)
{
    long long i = blockIdx.x * 256LL + threadIdx.x;
    const long long tot = (long long)Nb * Ss * Dm;
    if (i >= tot) return;
    const long long r = i / Dm;
    const int s = (int)(r % Ss);
    const long long b = r / Ss;
    const float m = (s == 0) ? 1.f : pm[b * Ll + (s - 1)];
    out[i] = x2[i] * m * 2.f + x[i];
}

// ---------------------------------------------------------------------------
extern "C" void kernel_launch(void* const* d_in, const int* in_sizes, int n_in,
                              void* d_out, int out_size, void* d_ws, size_t ws_size,
                              hipStream_t stream)
{
    (void)in_sizes; (void)n_in; (void)out_size; (void)ws_size;

    const float* masked_img = (const float*)d_in[0];
    const float* prompt_img = (const float*)d_in[1];
    const float* pred_masks = (const float*)d_in[2];
    const float* conv_w     = (const float*)d_in[3];
    const float* class_emb  = (const float*)d_in[4];
    const float* pos_emb    = (const float*)d_in[5];
    const float* ln_pre_g   = (const float*)d_in[6];
    const float* ln_pre_b   = (const float*)d_in[7];
    const float* Wqkv       = (const float*)d_in[8];
    const float* bqkv       = (const float*)d_in[9];
    const float* Wo         = (const float*)d_in[10];
    const float* bo         = (const float*)d_in[11];
    const float* ln1_g      = (const float*)d_in[12];
    const float* ln1_b      = (const float*)d_in[13];
    const float* ln2_g      = (const float*)d_in[14];
    const float* ln2_b      = (const float*)d_in[15];
    const float* Wfc        = (const float*)d_in[16];
    const float* bfc        = (const float*)d_in[17];
    const float* Wp         = (const float*)d_in[18];
    const float* bp         = (const float*)d_in[19];
    const float* ln_post_g  = (const float*)d_in[20];
    const float* ln_post_b  = (const float*)d_in[21];
    const float* proj       = (const float*)d_in[22];

    // -------- workspace carve (256B aligned) --------
    char* wsp = (char*)d_ws;
    auto carve = [&](size_t bytes) -> char* {
        char* p = wsp;
        wsp += (bytes + 255) & ~(size_t)255;
        return p;
    };
    float*    x12    = (float*)   carve((size_t)BFULL * Ss * Dm * 4);
    float*    tmp    = (float*)   carve((size_t)Nb    * Ss * Dm * 4);
    _Float16* ln16   = (_Float16*)carve((size_t)BFULL * Ss * Dm * 2);
    _Float16* O16    = (_Float16*)carve((size_t)BFULL * Ss * Dm * 2);
    _Float16* qkv16  = (_Float16*)carve((size_t)BFULL * Ss * 3 * Dm * 2);
    _Float16* kT     = (_Float16*)carve((size_t)BFULL * Hh * HDh * KTP * 2);
    float*    scores = (float*)   carve((size_t)BFULL * Hh * Ss * SPp * 4);
    _Float16* probs  = (_Float16*)carve((size_t)BFULL * Hh * Ss * SPp * 2);
    _Float16* ffh16  = (_Float16*)carve((size_t)BFULL * Ss * FFm * 2);
    _Float16* wT     = (_Float16*)carve((size_t)FFm * Dm * 2);
    _Float16* cls16  = (_Float16*)carve((size_t)Nb * Dm * 2);

    auto blocks1d = [](long long tot) { return (unsigned)((tot + 255) / 256); };

    auto gemm = [&](const _Float16* A, long long lda,
                    const _Float16* B, long long ldb,
                    int M, int N, int K,
                    const float* bias, const float* resid,
                    float alpha, int act,
                    float* C32, long long ldc32, _Float16* C16, long long ldc16,
                    int Z, int hp,
                    long long sAb, long long sAh, long long sBb, long long sBh,
                    long long sC32b, long long sC32h, long long sC16b, long long sC16h) {
        dim3 g((N + 63) / 64, (M + 127) / 128, Z), blk(256);
        gemm_wmma_kernel<<<g, blk, 0, stream>>>(A, lda, B, ldb, M, N, K,
            bias, resid, (long long)Dm, alpha, act, C32, ldc32, C16, ldc16,
            hp, sAb, sAh, sBb, sBh, sC32b, sC32h, sC16b, sC16h);
    };
    auto gemm_plain = [&](const _Float16* A, int lda, const _Float16* B, int ldb,
                          int M, int N, int K, const float* bias, const float* resid,
                          float alpha, int act,
                          float* C32, int ldc32, _Float16* C16, int ldc16) {
        gemm(A, lda, B, ldb, M, N, K, bias, resid, alpha, act,
             C32, ldc32, C16, ldc16, 1, 1, 0, 0, 0, 0, 0, 0, 0, 0);
    };

    // -------- one transformer block (in place on X) --------
    auto run_block = [&](float* X, int Bb, int layer, const float* pmMask) {
        const int M = Bb * Ss;
        // ln1 -> f16
        ln_kernel<<<M, 256, 0, stream>>>(X, Dm, M, Dm,
            ln1_g + (long long)layer * Dm, ln1_b + (long long)layer * Dm,
            nullptr, 0, ln16, Dm);
        // qkv = ln1(x) @ Wqkv^T + bqkv   (Wqkv converted transposed -> NN form)
        convert_tr_kernel<<<blocks1d((long long)3 * Dm * Dm), 256, 0, stream>>>(
            Wqkv + (long long)layer * 3 * Dm * Dm, wT, 3 * Dm, Dm);
        gemm_plain(ln16, Dm, wT, 3 * Dm, M, 3 * Dm, Dm,
                   bqkv + (long long)layer * 3 * Dm, nullptr, 1.f, 0,
                   nullptr, 0, qkv16, 3 * Dm);
        // K^T tiles (padded to 256 cols), then scores = Q K^T / 8 (NN form)
        kt_fill_kernel<<<blocks1d((long long)Bb * Hh * HDh * KTP), 256, 0, stream>>>(
            qkv16, kT, Bb);
        gemm(qkv16, 3 * Dm, kT, KTP, Ss, Ss, HDh,
             nullptr, nullptr, 0.125f, 0, scores, SPp, nullptr, 0,
             Bb * Hh, Hh,
             (long long)Ss * 3 * Dm, HDh,
             (long long)Hh * HDh * KTP, (long long)HDh * KTP,
             (long long)Hh * Ss * SPp, (long long)Ss * SPp, 0, 0);
        // softmax (+ CLS mask bias on the x2 path)
        softmax_kernel<<<dim3(Ss, Bb * Hh), 256, 0, stream>>>(scores, probs, pmMask, Hh);
        // O = P V  (NN form, K padded to 224, pad probs are zero)
        gemm(probs, SPp, qkv16 + 2 * Dm, 3 * Dm, Ss, HDh, SPp,
             nullptr, nullptr, 1.f, 0, nullptr, 0, O16, Dm,
             Bb * Hh, Hh,
             (long long)Hh * Ss * SPp, (long long)Ss * SPp,
             (long long)Ss * 3 * Dm, HDh,
             0, 0, (long long)Ss * Dm, HDh);
        // x += O @ Wo^T + bo
        convert_tr_kernel<<<blocks1d((long long)Dm * Dm), 256, 0, stream>>>(
            Wo + (long long)layer * Dm * Dm, wT, Dm, Dm);
        gemm_plain(O16, Dm, wT, Dm, M, Dm, Dm,
                   bo + (long long)layer * Dm, X, 1.f, 0, X, Dm, nullptr, 0);
        // h = act(ln2(x) @ Wfc^T + bfc)  -> f16
        ln_kernel<<<M, 256, 0, stream>>>(X, Dm, M, Dm,
            ln2_g + (long long)layer * Dm, ln2_b + (long long)layer * Dm,
            nullptr, 0, ln16, Dm);
        convert_tr_kernel<<<blocks1d((long long)FFm * Dm), 256, 0, stream>>>(
            Wfc + (long long)layer * FFm * Dm, wT, FFm, Dm);
        gemm_plain(ln16, Dm, wT, FFm, M, FFm, Dm,
                   bfc + (long long)layer * FFm, nullptr, 1.f, 1,
                   nullptr, 0, ffh16, FFm);
        // x += h @ Wp^T + bp
        convert_tr_kernel<<<blocks1d((long long)Dm * FFm), 256, 0, stream>>>(
            Wp + (long long)layer * Dm * FFm, wT, Dm, FFm);
        gemm_plain(ffh16, FFm, wT, Dm, M, Dm, FFm,
                   bp + (long long)layer * Dm, X, 1.f, 0, X, Dm, nullptr, 0);
    };

    // -------- patch embedding (both image sets), conv as GEMM --------
    im2col_kernel<<<blocks1d((long long)BFULL * Ll * Dm), 256, 0, stream>>>(
        masked_img, prompt_img, ln16);
    convert_tr_kernel<<<blocks1d((long long)Dm * Dm), 256, 0, stream>>>(
        conv_w, wT, Dm, Dm);
    float* feat = scores;   // scratch reuse
    gemm_plain(ln16, Dm, wT, Dm, BFULL * Ll, Dm, Dm,
               nullptr, nullptr, 1.f, 0, feat, Dm, nullptr, 0);
    float* xraw = (float*)ffh16;  // scratch reuse
    assemble_kernel<<<blocks1d((long long)BFULL * Ss * Dm), 256, 0, stream>>>(
        feat, class_emb, pos_emb, xraw);
    ln_kernel<<<BFULL * Ss, 256, 0, stream>>>(xraw, Dm, BFULL * Ss, Dm,
        ln_pre_g, ln_pre_b, x12, Dm, nullptr, 0);

    // -------- layers 0..9 on concatenated batch --------
    for (int i = 0; i < 10; ++i) run_block(x12, BFULL, i, nullptr);

    // -------- masked layers 10, 11 on split batches --------
    float* x  = x12;
    float* x2 = x12 + (long long)Nb * Ss * Dm;
    const int maskedLayers[2] = {10, 11};
    for (int t = 0; t < 2; ++t) {
        const int layer = maskedLayers[t];
        masked_combine_kernel<<<blocks1d((long long)Nb * Ss * Dm), 256, 0, stream>>>(
            tmp, x, x2, pred_masks);
        run_block(tmp, Nb, layer, nullptr);          // x-path (uses old x2 via tmp)
        run_block(x2, Nb, layer, pred_masks);        // x2-path with CLS-row mask bias
        hipMemcpyAsync(x, tmp, (size_t)Nb * Ss * Dm * sizeof(float),
                       hipMemcpyDeviceToDevice, stream);
    }

    // -------- final: ln_post(cls) @ proj --------
    ln_kernel<<<Nb, 256, 0, stream>>>(x, (long long)Ss * Dm, Nb, Dm,
        ln_post_g, ln_post_b, nullptr, 0, cls16, Dm);
    convert_kernel<<<blocks1d((long long)Dm * OUTm), 256, 0, stream>>>(
        proj, wT, (long long)Dm * OUTm);            // proj is already [K=768, N=512]
    gemm_plain(cls16, Dm, wT, OUTm, Nb, OUTm, Dm,
               nullptr, nullptr, 1.f, 0, (float*)d_out, OUTm, nullptr, 0);
}
